// AttentionLayer_45629732552708
// MI455X (gfx1250) — compile-verified, matrix operand order checked by reading, not run on hardware
//
#include <hip/hip_runtime.h>
#include <hip/hip_bf16.h>
#include <stdint.h>

// ---- CDNA5 WMMA vector types ----
typedef __attribute__((ext_vector_type(16))) __bf16   v16bf;
typedef __attribute__((ext_vector_type(8)))  float    v8f;
typedef __attribute__((ext_vector_type(4)))  uint32_t u32x4;

#define BATCH 8
#define TQ    2048
#define TK    2048
#define DIM   1024
#define DVDIM 1024
#define KB    128            // keys processed per outer iteration
#define QROWS 16             // q rows per block
#define QSTR  (DIM + 8)      // LDS stride (bf16 elems) for Q tile: 2064B rows -> bank-spread
#define PSTR  (KB + 8)       // LDS stride (bf16 elems) for P tile: 272B rows -> bank-spread

// Pack two f32 -> packed bf16 pair (truncate) in ONE v_perm_b32:
// result[15:0] = lo[31:16], result[31:16] = hi[31:16]
static __device__ __forceinline__ uint32_t pk_bf16(float lo, float hi) {
  return __builtin_amdgcn_perm(__builtin_bit_cast(uint32_t, hi),
                               __builtin_bit_cast(uint32_t, lo),
                               0x07060302u);
}

static __device__ __forceinline__ __bf16 bf16_of(float x) {
  uint32_t u = __builtin_bit_cast(uint32_t, x);
  unsigned short h = (unsigned short)(u >> 16);
  return __builtin_bit_cast(__bf16, h);
}

union ABFrag { v16bf v; u32x4 q[2]; uint32_t u[8]; };

__global__ __launch_bounds__(256)
void attn_tanh_softmax_kernel(const float* __restrict__ Q,
                              const float* __restrict__ K,
                              const float* __restrict__ V,
                              const float* __restrict__ Bias,
                              float* __restrict__ Out) {
  __shared__ __align__(16) __bf16 sQ[QROWS * QSTR];   // Q tile, bf16, row-major padded
  __shared__ __align__(16) __bf16 sP[QROWS * PSTR];   // P tile (exp(tanh)) for current key block

  const int tid = threadIdx.x;
  const int w  = tid >> 5;        // wave id 0..7
  const int l  = tid & 31;        // lane
  const int ln = l & 15;          // lane % 16 (N / M index inside fragment)
  const int lh = l >> 4;          // half-wave select
  const int blk   = blockIdx.x;
  const int batch = blk >> 7;           // 128 q-tiles per batch
  const int q0    = (blk & 127) * QROWS;

  // ---------- preload Q tile (16 x 1024 f32 -> bf16 in LDS) ----------
  {
    const int r  = tid >> 4;            // row 0..15
    const int c0 = (tid & 15) * 64;     // 64 consecutive cols per thread
    const float* qrow = Q + ((size_t)batch * TQ + (size_t)(q0 + r)) * DIM + c0;
    uint32_t* dst = (uint32_t*)&sQ[r * QSTR + c0];
#pragma unroll
    for (int j = 0; j < 64; j += 2)
      dst[j >> 1] = pk_bf16(qrow[j], qrow[j + 1]);
  }
  __syncthreads();

  const v8f vzero = {0.f,0.f,0.f,0.f,0.f,0.f,0.f,0.f};
  v8f accV[8];
#pragma unroll
  for (int i = 0; i < 8; ++i) accV[i] = vzero;
  v8f accSum = vzero;                    // softmax denominator via P x ones (WMMA)

  ABFrag ones;                           // B matrix of all 1.0bf16 (0x3F80)
#pragma unroll
  for (int e = 0; e < 8; ++e) ones.u[e] = 0x3F803F80u;

  const float* kbat = K + (size_t)batch * DIM * TK;    // K[d, j]
  const float* vbat = V + (size_t)batch * TK * DVDIM;  // V[j, n]

#pragma unroll 1
  for (int j0 = 0; j0 < TK; j0 += KB) {
    // ================= QK phase: wave w -> S tile for 16 keys [j0 + 16w, +16) ======
    const int   key  = j0 + w * 16 + ln;       // this lane's key column (N index)
    const float bias = Bias[key];
    const float* kb  = kbat + key;             // + d*TK walks the d axis
    v8f accS = vzero;
#pragma unroll 2
    for (int dk = 0; dk < DIM; dk += 32) {
      ABFrag a, b;
      // A fragment: Q rows from LDS (elems 0..7 = d0..d0+7 (+8 hi half), 8..15 = d0+16..23)
      const __bf16* qp = &sQ[ln * QSTR + dk + lh * 8];
      a.q[0] = *(const u32x4*)(qp);
      a.q[1] = *(const u32x4*)(qp + 16);
      // B fragment: K[d, key] gathered along d (lanes 0..15 hit one 64B line per element)
      const float* kp = kb + (size_t)(dk + lh * 8) * TK;
      float f[16];
#pragma unroll
      for (int e = 0; e < 8; ++e) {
        f[e]     = kp[(size_t)e * TK];
        f[e + 8] = kp[(size_t)(e + 16) * TK];
      }
#pragma unroll
      for (int e = 0; e < 8; ++e) b.u[e] = pk_bf16(f[2 * e], f[2 * e + 1]);
      accS = __builtin_amdgcn_wmma_f32_16x16x32_bf16(false, a.v, false, b.v,
                                                     (short)0, accS, false, false);
    }
    // scores -> p = exp(tanh(s + bias)); tanh in (-1,1) => exp is safe, no running max.
#pragma unroll
    for (int r = 0; r < 8; ++r) {
      const int m = r + lh * 8;                       // C/D layout: VGPR r, half lh -> row m
      float s  = accS[r] + bias;
      float e2 = __builtin_amdgcn_exp2f(s * 2.885390082f);          // exp(2s)
      float th = 1.0f - 2.0f * __builtin_amdgcn_rcpf(1.0f + e2);    // tanh(s)
      float p  = __builtin_amdgcn_exp2f(th * 1.442695041f);         // exp(tanh(s))
      sP[m * PSTR + (w * 16 + ln)] = bf16_of(p);
    }
    __syncthreads();

    // ================= PV phase: wave w -> DV columns [128w, 128w + 128) ==========
    // Prefetch next key block's K/V lines while the WMMA pipe fills.
    if (j0 + KB < TK) {
      __builtin_prefetch(kbat + (size_t)(j0 + KB) + w * 16 + ln, 0, 3);
      __builtin_prefetch(vbat + (size_t)(j0 + KB + lh * 8) * DVDIM + w * 128 + ln, 0, 3);
    }
    ABFrag pA[4];                                     // P A-fragments for 4 k-steps
#pragma unroll
    for (int s4 = 0; s4 < 4; ++s4) {
      const __bf16* pp = &sP[ln * PSTR + s4 * 32 + lh * 8];
      pA[s4].q[0] = *(const u32x4*)(pp);
      pA[s4].q[1] = *(const u32x4*)(pp + 16);
    }
    // Softmax denominator: accSum += P(16x128) x ones(128x16); every lane of row m
    // ends up holding sum_j P[m,j]. Replaces ds_bpermute reductions + LDS atomics.
#pragma unroll
    for (int s4 = 0; s4 < 4; ++s4)
      accSum = __builtin_amdgcn_wmma_f32_16x16x32_bf16(false, pA[s4].v, false, ones.v,
                                                       (short)0, accSum, false, false);
    const float* vb = vbat + (size_t)(j0 + lh * 8) * DVDIM + w * 128 + ln;
#pragma unroll 1
    for (int nt = 0; nt < 8; ++nt) {
      const float* vn = vb + nt * 16;
#pragma unroll
      for (int s4 = 0; s4 < 4; ++s4) {
        const float* vp = vn + (size_t)(s4 * 32) * DVDIM;
        ABFrag b;
        float f[16];
#pragma unroll
        for (int e = 0; e < 8; ++e) {
          f[e]     = vp[(size_t)e * DVDIM];
          f[e + 8] = vp[(size_t)(e + 16) * DVDIM];
        }
#pragma unroll
        for (int e = 0; e < 8; ++e) b.u[e] = pk_bf16(f[2 * e], f[2 * e + 1]);
        accV[nt] = __builtin_amdgcn_wmma_f32_16x16x32_bf16(false, pA[s4].v, false, b.v,
                                                           (short)0, accV[nt], false, false);
      }
    }
    __syncthreads();
  }

  // ---------- normalize & store ----------
  float inv[8];
#pragma unroll
  for (int r = 0; r < 8; ++r)
    inv[r] = __builtin_amdgcn_rcpf(accSum[r]);
  float* ob = Out + ((size_t)batch * TQ + q0) * DVDIM + w * 128 + ln;
#pragma unroll
  for (int nt = 0; nt < 8; ++nt) {
#pragma unroll
    for (int r = 0; r < 8; ++r)
      ob[(size_t)(r + lh * 8) * DVDIM + nt * 16] = accV[nt][r] * inv[r];
  }
}

extern "C" void kernel_launch(void* const* d_in, const int* in_sizes, int n_in,
                              void* d_out, int out_size, void* d_ws, size_t ws_size,
                              hipStream_t stream) {
  (void)in_sizes; (void)n_in; (void)out_size; (void)d_ws; (void)ws_size;
  const float* q = (const float*)d_in[0];
  const float* k = (const float*)d_in[1];
  const float* v = (const float*)d_in[2];
  const float* b = (const float*)d_in[3];
  float* out = (float*)d_out;
  dim3 grid(BATCH * (TQ / QROWS));   // 1024 blocks: (batch, 16-row q tile)
  dim3 block(256);                   // 8 wave32 waves
  attn_tanh_softmax_kernel<<<grid, block, 0, stream>>>(q, k, v, b, out);
}
